// _MedformerLayer_55774445306391
// MI455X (gfx1250) — compile-verified, matrix operand order checked by reading, not run on hardware
//
#include <hip/hip_runtime.h>

// ---------------------------------------------------------------------------
// Medformer layer for MI455X (gfx1250): all GEMMs + attention via
// v_wmma_f32_16x16x32_f16 (wave32). f16 operands, f32 accumulation.
// Big GEMMs use 64x128 block tiles (wave tile 32x64 -> 8 WMMA / 12 ds_load_b128
// per K-step) + global_prefetch_b8 for the next K-step tiles.
// ---------------------------------------------------------------------------

#define NB_ 4
#define B_  16
#define L_  512
#define D_  512
#define H_  8
#define DH_ 64
// SCALE * log2(e) = (1/8) * 1.4426950408889634
#define SCALE_LOG2E 0.18033688011112043f

typedef __attribute__((ext_vector_type(16))) _Float16 v16h;
typedef __attribute__((ext_vector_type(8)))  float    v8f;
typedef __attribute__((ext_vector_type(4)))  _Float16 h4;

union FragU { uint4 u[2]; v16h v; };

// Load a 16x32 f16 fragment (A-matrix layout, or B^T stored row-major [n][k]).
// Documented layout: lane l holds row m=l&15; lanes 0-15 hold k = 0..7 & 16..23,
// lanes 16-31 hold k = 8..15 & 24..31 -> two contiguous 16-byte loads per lane.
__device__ __forceinline__ v16h load_frag16(const _Float16* base, int row_stride, int lane) {
    const int m  = lane & 15;
    const int kb = (lane >> 4) << 3;
    const _Float16* p = base + (size_t)m * row_stride;
    FragU f;
    f.u[0] = *(const uint4*)(p + kb);
    f.u[1] = *(const uint4*)(p + 16 + kb);
    return f.v;
}

__device__ __forceinline__ v8f wmma16(v16h a, v16h b, v8f c) {
    return __builtin_amdgcn_wmma_f32_16x16x32_f16(false, a, false, b, (short)0, c, false, false);
}

// ---------------------------------------------------------------------------
// Kernel 1: intra QKV projection. out = X @ W + b, output f16.
// grid = (128*4, 3, NB), block = 128 (4 waves). 64x128 tile, BK=32.
// Wave tile 32x64: acc[2][4], 8 WMMA per K-step.
// ---------------------------------------------------------------------------
__global__ __launch_bounds__(128)
void k_gemm_qkv_intra(const float* __restrict__ X,
                      const float* __restrict__ Wq, const float* __restrict__ bq,
                      const float* __restrict__ Wk, const float* __restrict__ bk,
                      const float* __restrict__ Wv, const float* __restrict__ bv,
                      _Float16* __restrict__ Qb, _Float16* __restrict__ Kb,
                      _Float16* __restrict__ Vb) {
    __shared__ _Float16 As[64 * 32];    // [m][k]
    __shared__ _Float16 Bs[128 * 32];   // [n][k]  (W transposed tile)
    const int nb  = blockIdx.z;
    const int sel = blockIdx.y;
    const int mt  = blockIdx.x >> 2;    // M/64 = 128 tiles
    const int nt  = blockIdx.x & 3;     // N/128 = 4 tiles
    const int mbase = mt * 64, nbase = nt * 128;

    const float* W    = (sel == 0 ? Wq : sel == 1 ? Wk : Wv) + (size_t)nb * D_ * D_;
    const float* bias = (sel == 0 ? bq : sel == 1 ? bk : bv) + (size_t)nb * D_;
    _Float16*    Out  = (sel == 0 ? Qb : sel == 1 ? Kb : Vb) + (size_t)nb * (B_ * L_) * D_;
    const float* Xp   = X + (size_t)nb * (B_ * L_) * D_;

    const int tid = threadIdx.x;
    const int wave = tid >> 5, lane = tid & 31;
    const int hi = lane >> 4, ln = lane & 15;
    const int wm = (wave >> 1) * 32;    // 0 or 32
    const int wn = (wave & 1) * 64;     // 0 or 64

    v8f acc[2][4];
#pragma unroll
    for (int mi = 0; mi < 2; ++mi)
#pragma unroll
        for (int ni = 0; ni < 4; ++ni) {
            float bvv = bias[nbase + wn + ni * 16 + ln];
#pragma unroll
            for (int j = 0; j < 8; ++j) acc[mi][ni][j] = bvv;
        }

    for (int kb = 0; kb < D_; kb += 32) {
        __syncthreads();
        {   // stage A (f32 -> f16), vectorized
            int m  = tid >> 1;
            int ks = (tid & 1) * 16;
            const float* xp = Xp + (size_t)(mbase + m) * D_ + kb + ks;
#pragma unroll
            for (int i = 0; i < 4; ++i) {
                float4 x4 = *(const float4*)(xp + 4 * i);
                h4 hv = { (_Float16)x4.x, (_Float16)x4.y, (_Float16)x4.z, (_Float16)x4.w };
                *(h4*)&As[m * 32 + ks + 4 * i] = hv;
            }
            if (kb + 32 < D_) __builtin_prefetch(xp + 32, 0, 3);
        }
        {   // stage B transposed: Bs[n][k] = W[kb+k][nbase+n], 128 cols
            int kk = tid >> 4;          // 0..7
            int n0 = (tid & 15) * 8;    // 0..120
#pragma unroll
            for (int jj = 0; jj < 4; ++jj) {
                int k = kk + 8 * jj;
                const float* wp = W + (size_t)(kb + k) * D_ + nbase + n0;
                float4 w4 = *(const float4*)(wp);
                float4 w5 = *(const float4*)(wp + 4);
                Bs[(n0 + 0) * 32 + k] = (_Float16)w4.x;
                Bs[(n0 + 1) * 32 + k] = (_Float16)w4.y;
                Bs[(n0 + 2) * 32 + k] = (_Float16)w4.z;
                Bs[(n0 + 3) * 32 + k] = (_Float16)w4.w;
                Bs[(n0 + 4) * 32 + k] = (_Float16)w5.x;
                Bs[(n0 + 5) * 32 + k] = (_Float16)w5.y;
                Bs[(n0 + 6) * 32 + k] = (_Float16)w5.z;
                Bs[(n0 + 7) * 32 + k] = (_Float16)w5.w;
                if (kb + 32 < D_) __builtin_prefetch(wp + 32 * D_, 0, 3);
            }
        }
        __syncthreads();
        v16h a0 = load_frag16(&As[(wm)      * 32], 32, lane);
        v16h a1 = load_frag16(&As[(wm + 16) * 32], 32, lane);
#pragma unroll
        for (int ni = 0; ni < 4; ++ni) {
            v16h b = load_frag16(&Bs[(wn + ni * 16) * 32], 32, lane);
            acc[0][ni] = wmma16(a0, b, acc[0][ni]);
            acc[1][ni] = wmma16(a1, b, acc[1][ni]);
        }
    }

#pragma unroll
    for (int mi = 0; mi < 2; ++mi)
#pragma unroll
        for (int ni = 0; ni < 4; ++ni)
#pragma unroll
            for (int j = 0; j < 8; ++j) {
                int row = mbase + wm + mi * 16 + j + 8 * hi;
                int col = nbase + wn + ni * 16 + ln;
                Out[(size_t)row * D_ + col] = (_Float16)acc[mi][ni][j];
            }
}

// ---------------------------------------------------------------------------
// Kernel 2: flash attention per (nb,b,h). grid = (L/64, H, NB*B), block 128.
// Each wave owns 16 Q rows; loops 32-key chunks with online softmax.
// Writes output over the Q buffer (safe: each row's Q read only by its owner).
// ---------------------------------------------------------------------------
__global__ __launch_bounds__(128)
void k_flash(const _Float16* Q, const _Float16* __restrict__ K,
             const _Float16* __restrict__ V, _Float16* Obuf) {
    __shared__ _Float16 Vt[64 * 32];        // [feat][key] transposed V chunk
    __shared__ _Float16 Plds[4][16 * 32];   // per-wave P re-layout buffer

    const int mtile = blockIdx.x;
    const int h     = blockIdx.y;
    const int bz    = blockIdx.z;           // nb*B + b
    const int tid   = threadIdx.x;
    const int wave  = tid >> 5, lane = tid & 31;
    const int hi = lane >> 4, ln = lane & 15;

    const size_t headoff = (size_t)bz * L_ * D_ + (size_t)h * DH_;
    const _Float16* Qp = Q + headoff;
    const _Float16* Kp = K + headoff;
    const _Float16* Vp = V + headoff;

    const int m0 = mtile * 64 + wave * 16;

    v16h qa0 = load_frag16(Qp + (size_t)m0 * D_,      D_, lane);
    v16h qa1 = load_frag16(Qp + (size_t)m0 * D_ + 32, D_, lane);

    v8f o0 = {}, o1 = {}, o2 = {}, o3 = {};
    float rmax[8], rsum[8];
#pragma unroll
    for (int j = 0; j < 8; ++j) { rmax[j] = -1e30f; rsum[j] = 0.f; }

    for (int kc = 0; kc < L_ / 32; ++kc) {
        const int key0 = kc * 32;
        __syncthreads();
        {   // cooperative stage: Vt[feat][key] = V[key0+key][feat]
            int key = tid & 31;
            int fg  = tid >> 5;   // 0..3 -> feats fg*16..fg*16+15
            const _Float16* vp = Vp + (size_t)(key0 + key) * D_ + fg * 16;
            uint4 va = *(const uint4*)(vp);
            uint4 vb = *(const uint4*)(vp + 8);
            const _Float16* va_h = (const _Float16*)&va;
            const _Float16* vb_h = (const _Float16*)&vb;
#pragma unroll
            for (int i = 0; i < 8; ++i) {
                Vt[(fg * 16 + i)     * 32 + key] = va_h[i];
                Vt[(fg * 16 + 8 + i) * 32 + key] = vb_h[i];
            }
            if (key0 + 32 < L_) __builtin_prefetch(vp + (size_t)32 * D_, 0, 3);
        }
        __syncthreads();

        // S = Q @ K^T for 2 key sub-tiles; K fragments straight from global
        v8f s0 = {}, s1 = {};
        {
            const int kb = hi * 8;
            const _Float16* kp0 = Kp + (size_t)(key0 + ln)      * D_;
            const _Float16* kp1 = Kp + (size_t)(key0 + 16 + ln) * D_;
            FragU b;
            b.u[0] = *(const uint4*)(kp0 + kb);
            b.u[1] = *(const uint4*)(kp0 + 16 + kb);
            s0 = wmma16(qa0, b.v, s0);
            b.u[0] = *(const uint4*)(kp0 + 32 + kb);
            b.u[1] = *(const uint4*)(kp0 + 48 + kb);
            s0 = wmma16(qa1, b.v, s0);
            b.u[0] = *(const uint4*)(kp1 + kb);
            b.u[1] = *(const uint4*)(kp1 + 16 + kb);
            s1 = wmma16(qa0, b.v, s1);
            b.u[0] = *(const uint4*)(kp1 + 32 + kb);
            b.u[1] = *(const uint4*)(kp1 + 48 + kb);
            s1 = wmma16(qa1, b.v, s1);
            if (key0 + 32 < L_) __builtin_prefetch(kp0 + (size_t)32 * D_, 0, 3);
        }

        // online softmax (row m = j + 8*hi lives across the lane's 16-group)
        float p0[8], p1[8];
#pragma unroll
        for (int j = 0; j < 8; ++j) {
            float cm = fmaxf(s0[j], s1[j]);
#pragma unroll
            for (int msk = 1; msk < 16; msk <<= 1)
                cm = fmaxf(cm, __shfl_xor(cm, msk, 32));
            float nm    = fmaxf(rmax[j], cm);
            float alpha = exp2f((rmax[j] - nm) * SCALE_LOG2E);
            rmax[j] = nm;
            float e0 = exp2f((s0[j] - nm) * SCALE_LOG2E);
            float e1 = exp2f((s1[j] - nm) * SCALE_LOG2E);
            float ps = e0 + e1;
#pragma unroll
            for (int msk = 1; msk < 16; msk <<= 1)
                ps += __shfl_xor(ps, msk, 32);
            rsum[j] = rsum[j] * alpha + ps;
            o0[j] *= alpha; o1[j] *= alpha; o2[j] *= alpha; o3[j] *= alpha;
            p0[j] = e0; p1[j] = e1;
        }

        // C-layout -> A-layout round trip through per-wave LDS
        _Float16* pl = &Plds[wave][0];
#pragma unroll
        for (int j = 0; j < 8; ++j) {
            int row = j + 8 * hi;
            pl[row * 32 + ln]      = (_Float16)p0[j];
            pl[row * 32 + 16 + ln] = (_Float16)p1[j];
        }
        v16h pa = load_frag16(pl, 32, lane);
        o0 = wmma16(pa, load_frag16(&Vt[0 * 16 * 32], 32, lane), o0);
        o1 = wmma16(pa, load_frag16(&Vt[1 * 16 * 32], 32, lane), o1);
        o2 = wmma16(pa, load_frag16(&Vt[2 * 16 * 32], 32, lane), o2);
        o3 = wmma16(pa, load_frag16(&Vt[3 * 16 * 32], 32, lane), o3);
    }

    // finalize + store (over the Q buffer region)
#pragma unroll
    for (int j = 0; j < 8; ++j) {
        float inv = 1.0f / rsum[j];
        int row = j + 8 * hi;
        _Float16* orow = Obuf + headoff + (size_t)(m0 + row) * D_;
        orow[ln]      = (_Float16)(o0[j] * inv);
        orow[16 + ln] = (_Float16)(o1[j] * inv);
        orow[32 + ln] = (_Float16)(o2[j] * inv);
        orow[48 + ln] = (_Float16)(o3[j] * inv);
    }
}

// ---------------------------------------------------------------------------
// Kernel 3: intra output projection. out(f32) = attn(f16) @ Wo + bo.
// grid = (128*4, 1, NB), block 128. 64x128 tile.
// ---------------------------------------------------------------------------
__global__ __launch_bounds__(128)
void k_gemm_oproj_intra(const _Float16* __restrict__ A,
                        const float* __restrict__ Wo, const float* __restrict__ bo,
                        float* __restrict__ Outf) {
    __shared__ _Float16 As[64 * 32];
    __shared__ _Float16 Bs[128 * 32];
    const int nb = blockIdx.z;
    const int mt = blockIdx.x >> 2;
    const int nt = blockIdx.x & 3;
    const int mbase = mt * 64, nbase = nt * 128;

    const float* W    = Wo + (size_t)nb * D_ * D_;
    const float* bias = bo + (size_t)nb * D_;
    const _Float16* Ap = A + (size_t)nb * (B_ * L_) * D_;
    float* Op = Outf + (size_t)nb * (B_ * L_) * D_;

    const int tid = threadIdx.x;
    const int wave = tid >> 5, lane = tid & 31;
    const int hi = lane >> 4, ln = lane & 15;
    const int wm = (wave >> 1) * 32;
    const int wn = (wave & 1) * 64;

    v8f acc[2][4];
#pragma unroll
    for (int mi = 0; mi < 2; ++mi)
#pragma unroll
        for (int ni = 0; ni < 4; ++ni) {
            float bvv = bias[nbase + wn + ni * 16 + ln];
#pragma unroll
            for (int j = 0; j < 8; ++j) acc[mi][ni][j] = bvv;
        }

    for (int kb = 0; kb < D_; kb += 32) {
        __syncthreads();
        {   // stage A (already f16): two 16-byte copies per thread
            int m  = tid >> 1;
            int ks = (tid & 1) * 16;
            const _Float16* ap = Ap + (size_t)(mbase + m) * D_ + kb + ks;
            *(uint4*)&As[m * 32 + ks]     = *(const uint4*)(ap);
            *(uint4*)&As[m * 32 + ks + 8] = *(const uint4*)(ap + 8);
            if (kb + 32 < D_) __builtin_prefetch(ap + 32, 0, 3);
        }
        {   // stage B transposed, 128 cols
            int kk = tid >> 4;
            int n0 = (tid & 15) * 8;
#pragma unroll
            for (int jj = 0; jj < 4; ++jj) {
                int k = kk + 8 * jj;
                const float* wp = W + (size_t)(kb + k) * D_ + nbase + n0;
                float4 w4 = *(const float4*)(wp);
                float4 w5 = *(const float4*)(wp + 4);
                Bs[(n0 + 0) * 32 + k] = (_Float16)w4.x;
                Bs[(n0 + 1) * 32 + k] = (_Float16)w4.y;
                Bs[(n0 + 2) * 32 + k] = (_Float16)w4.z;
                Bs[(n0 + 3) * 32 + k] = (_Float16)w4.w;
                Bs[(n0 + 4) * 32 + k] = (_Float16)w5.x;
                Bs[(n0 + 5) * 32 + k] = (_Float16)w5.y;
                Bs[(n0 + 6) * 32 + k] = (_Float16)w5.z;
                Bs[(n0 + 7) * 32 + k] = (_Float16)w5.w;
                if (kb + 32 < D_) __builtin_prefetch(wp + 32 * D_, 0, 3);
            }
        }
        __syncthreads();
        v16h a0 = load_frag16(&As[(wm)      * 32], 32, lane);
        v16h a1 = load_frag16(&As[(wm + 16) * 32], 32, lane);
#pragma unroll
        for (int ni = 0; ni < 4; ++ni) {
            v16h b = load_frag16(&Bs[(wn + ni * 16) * 32], 32, lane);
            acc[0][ni] = wmma16(a0, b, acc[0][ni]);
            acc[1][ni] = wmma16(a1, b, acc[1][ni]);
        }
    }

#pragma unroll
    for (int mi = 0; mi < 2; ++mi)
#pragma unroll
        for (int ni = 0; ni < 4; ++ni)
#pragma unroll
            for (int j = 0; j < 8; ++j) {
                int row = mbase + wm + mi * 16 + j + 8 * hi;
                int col = nbase + wn + ni * 16 + ln;
                Op[(size_t)row * D_ + col] = acc[mi][ni][j];
            }
}

// ---------------------------------------------------------------------------
// Kernel 4: inter (router) QKV projection. M=64 rows = routers[b][nb] gathered
// from d_out last tokens. grid = (8, 3, 1), block 128. 64x64 tile.
// ---------------------------------------------------------------------------
__global__ __launch_bounds__(128)
void k_gemm_qkv_inter(const float* __restrict__ Outf,
                      const float* __restrict__ iWq, const float* __restrict__ ibq,
                      const float* __restrict__ iWk, const float* __restrict__ ibk,
                      const float* __restrict__ iWv, const float* __restrict__ ibv,
                      _Float16* __restrict__ q2, _Float16* __restrict__ k2,
                      _Float16* __restrict__ v2) {
    __shared__ _Float16 As[64 * 32];
    __shared__ _Float16 Bs[64 * 32];
    const int sel = blockIdx.y;
    const int nbase = blockIdx.x * 64;

    const float* W    = (sel == 0 ? iWq : sel == 1 ? iWk : iWv);
    const float* bias = (sel == 0 ? ibq : sel == 1 ? ibk : ibv);
    _Float16*    Out  = (sel == 0 ? q2  : sel == 1 ? k2  : v2);

    const int tid = threadIdx.x;
    const int wave = tid >> 5, lane = tid & 31;
    const int hi = lane >> 4, ln = lane & 15;
    const int wm = (wave >> 1) * 32, wn = (wave & 1) * 32;

    v8f acc[2][2];
#pragma unroll
    for (int mi = 0; mi < 2; ++mi)
#pragma unroll
        for (int ni = 0; ni < 2; ++ni) {
            float bvv = bias[nbase + wn + ni * 16 + ln];
#pragma unroll
            for (int j = 0; j < 8; ++j) acc[mi][ni][j] = bvv;
        }

    for (int kb = 0; kb < D_; kb += 32) {
        __syncthreads();
        {   // stage A: row r=b*4+nb -> x_intra[nb][b][L-1][:] from d_out
            int r  = tid >> 1;
            int ks = (tid & 1) * 16;
            int nbb = r & 3, bb = r >> 2;
            const float* xp = Outf + (((size_t)(nbb * B_ + bb) * L_ + (L_ - 1)) * D_) + kb + ks;
#pragma unroll
            for (int i = 0; i < 4; ++i) {
                float4 x4 = *(const float4*)(xp + 4 * i);
                h4 hv = { (_Float16)x4.x, (_Float16)x4.y, (_Float16)x4.z, (_Float16)x4.w };
                *(h4*)&As[r * 32 + ks + 4 * i] = hv;
            }
        }
        {   // stage B transposed
            int kk = tid >> 4;
            int n0 = (tid & 15) * 4;
#pragma unroll
            for (int jj = 0; jj < 4; ++jj) {
                int k = kk + 8 * jj;
                float4 w4 = *(const float4*)(W + (size_t)(kb + k) * D_ + nbase + n0);
                Bs[(n0 + 0) * 32 + k] = (_Float16)w4.x;
                Bs[(n0 + 1) * 32 + k] = (_Float16)w4.y;
                Bs[(n0 + 2) * 32 + k] = (_Float16)w4.z;
                Bs[(n0 + 3) * 32 + k] = (_Float16)w4.w;
            }
        }
        __syncthreads();
        v16h a0 = load_frag16(&As[(wm)      * 32], 32, lane);
        v16h a1 = load_frag16(&As[(wm + 16) * 32], 32, lane);
        v16h b0 = load_frag16(&Bs[(wn)      * 32], 32, lane);
        v16h b1 = load_frag16(&Bs[(wn + 16) * 32], 32, lane);
        acc[0][0] = wmma16(a0, b0, acc[0][0]);
        acc[0][1] = wmma16(a0, b1, acc[0][1]);
        acc[1][0] = wmma16(a1, b0, acc[1][0]);
        acc[1][1] = wmma16(a1, b1, acc[1][1]);
    }

#pragma unroll
    for (int mi = 0; mi < 2; ++mi)
#pragma unroll
        for (int ni = 0; ni < 2; ++ni)
#pragma unroll
            for (int j = 0; j < 8; ++j) {
                int row = wm + mi * 16 + j + 8 * hi;
                int col = nbase + wn + ni * 16 + ln;
                Out[(size_t)row * D_ + col] = (_Float16)acc[mi][ni][j];
            }
}

// ---------------------------------------------------------------------------
// Kernel 5: tiny 4-token router attention (negligible FLOPs; scalar).
// One thread per (b,h). block 128, grid 1.
// ---------------------------------------------------------------------------
__global__ __launch_bounds__(128)
void k_inter_attn(const _Float16* __restrict__ q2, const _Float16* __restrict__ k2,
                  const _Float16* __restrict__ v2, _Float16* __restrict__ a2) {
    int t = threadIdx.x;
    if (t >= B_ * H_) return;
    int b = t >> 3, h = t & 7;
    float sc[NB_][NB_];
#pragma unroll
    for (int i = 0; i < NB_; ++i)
        for (int j = 0; j < NB_; ++j) {
            const _Float16* qp = q2 + (size_t)(b * NB_ + i) * D_ + h * DH_;
            const _Float16* kp = k2 + (size_t)(b * NB_ + j) * D_ + h * DH_;
            float s = 0.f;
            for (int f = 0; f < DH_; ++f) s += (float)qp[f] * (float)kp[f];
            sc[i][j] = s * 0.125f;
        }
#pragma unroll
    for (int i = 0; i < NB_; ++i) {
        float mx = fmaxf(fmaxf(sc[i][0], sc[i][1]), fmaxf(sc[i][2], sc[i][3]));
        float ssum = 0.f;
        for (int j = 0; j < NB_; ++j) { sc[i][j] = __expf(sc[i][j] - mx); ssum += sc[i][j]; }
        float inv = 1.f / ssum;
        for (int j = 0; j < NB_; ++j) sc[i][j] *= inv;
    }
    for (int i = 0; i < NB_; ++i) {
        _Float16* op = a2 + (size_t)(b * NB_ + i) * D_ + h * DH_;
        for (int f = 0; f < DH_; ++f) {
            float acc = 0.f;
            for (int j = 0; j < NB_; ++j)
                acc += sc[i][j] * (float)v2[(size_t)(b * NB_ + j) * D_ + h * DH_ + f];
            op[f] = (_Float16)acc;
        }
    }
}

// ---------------------------------------------------------------------------
// Kernel 6: inter output projection, scattered into d_out last tokens.
// grid = (8,1,1), block 128.
// ---------------------------------------------------------------------------
__global__ __launch_bounds__(128)
void k_gemm_oproj_inter(const _Float16* __restrict__ A,
                        const float* __restrict__ iWo, const float* __restrict__ ibo,
                        float* __restrict__ Outf) {
    __shared__ _Float16 As[64 * 32];
    __shared__ _Float16 Bs[64 * 32];
    const int nbase = blockIdx.x * 64;

    const int tid = threadIdx.x;
    const int wave = tid >> 5, lane = tid & 31;
    const int hi = lane >> 4, ln = lane & 15;
    const int wm = (wave >> 1) * 32, wn = (wave & 1) * 32;

    v8f acc[2][2];
#pragma unroll
    for (int mi = 0; mi < 2; ++mi)
#pragma unroll
        for (int ni = 0; ni < 2; ++ni) {
            float bvv = ibo[nbase + wn + ni * 16 + ln];
#pragma unroll
            for (int j = 0; j < 8; ++j) acc[mi][ni][j] = bvv;
        }

    for (int kb = 0; kb < D_; kb += 32) {
        __syncthreads();
        {
            int m  = tid >> 1;
            int ks = (tid & 1) * 16;
            const _Float16* ap = A + (size_t)m * D_ + kb + ks;
            *(uint4*)&As[m * 32 + ks]     = *(const uint4*)(ap);
            *(uint4*)&As[m * 32 + ks + 8] = *(const uint4*)(ap + 8);
        }
        {
            int kk = tid >> 4;
            int n0 = (tid & 15) * 4;
#pragma unroll
            for (int jj = 0; jj < 4; ++jj) {
                int k = kk + 8 * jj;
                float4 w4 = *(const float4*)(iWo + (size_t)(kb + k) * D_ + nbase + n0);
                Bs[(n0 + 0) * 32 + k] = (_Float16)w4.x;
                Bs[(n0 + 1) * 32 + k] = (_Float16)w4.y;
                Bs[(n0 + 2) * 32 + k] = (_Float16)w4.z;
                Bs[(n0 + 3) * 32 + k] = (_Float16)w4.w;
            }
        }
        __syncthreads();
        v16h a0 = load_frag16(&As[(wm)      * 32], 32, lane);
        v16h a1 = load_frag16(&As[(wm + 16) * 32], 32, lane);
        v16h b0 = load_frag16(&Bs[(wn)      * 32], 32, lane);
        v16h b1 = load_frag16(&Bs[(wn + 16) * 32], 32, lane);
        acc[0][0] = wmma16(a0, b0, acc[0][0]);
        acc[0][1] = wmma16(a0, b1, acc[0][1]);
        acc[1][0] = wmma16(a1, b0, acc[1][0]);
        acc[1][1] = wmma16(a1, b1, acc[1][1]);
    }

#pragma unroll
    for (int mi = 0; mi < 2; ++mi)
#pragma unroll
        for (int ni = 0; ni < 2; ++ni)
#pragma unroll
            for (int j = 0; j < 8; ++j) {
                int r   = wm + mi * 16 + j + 8 * hi;   // b*4+nb
                int col = nbase + wn + ni * 16 + ln;
                int nbb = r & 3, bb = r >> 2;
                Outf[(((size_t)(nbb * B_ + bb) * L_ + (L_ - 1)) * D_) + col] = acc[mi][ni][j];
            }
}

// ---------------------------------------------------------------------------
extern "C" void kernel_launch(void* const* d_in, const int* in_sizes, int n_in,
                              void* d_out, int out_size, void* d_ws, size_t ws_size,
                              hipStream_t stream) {
    const float* x   = (const float*)d_in[0];
    const float* Wq  = (const float*)d_in[1];
    const float* bq  = (const float*)d_in[2];
    const float* Wk  = (const float*)d_in[3];
    const float* bk  = (const float*)d_in[4];
    const float* Wv  = (const float*)d_in[5];
    const float* bv  = (const float*)d_in[6];
    const float* Wo  = (const float*)d_in[7];
    const float* bo  = (const float*)d_in[8];
    const float* iWq = (const float*)d_in[9];
    const float* ibq = (const float*)d_in[10];
    const float* iWk = (const float*)d_in[11];
    const float* ibk = (const float*)d_in[12];
    const float* iWv = (const float*)d_in[13];
    const float* ibv = (const float*)d_in[14];
    const float* iWo = (const float*)d_in[15];
    const float* ibo = (const float*)d_in[16];
    float* out = (float*)d_out;

    const size_t NBLD = (size_t)NB_ * B_ * L_ * D_;   // 16,777,216
    _Float16* ws = (_Float16*)d_ws;
    _Float16* Qb = ws;                 // reused as attn output after k_flash
    _Float16* Kb = ws + NBLD;
    _Float16* Vb = ws + 2 * NBLD;
    _Float16* q2 = ws + 3 * NBLD;
    _Float16* k2 = q2 + (size_t)B_ * NB_ * D_;
    _Float16* v2 = k2 + (size_t)B_ * NB_ * D_;
    _Float16* a2 = v2 + (size_t)B_ * NB_ * D_;

    // 1) intra QKV projection (64x128 tiles)
    dim3 g1((B_ * L_ / 64) * (D_ / 128), 3, NB_);
    k_gemm_qkv_intra<<<g1, 128, 0, stream>>>(x, Wq, bq, Wk, bk, Wv, bv, Qb, Kb, Vb);

    // 2) flash attention (writes attn over Qb)
    dim3 g2(L_ / 64, H_, NB_ * B_);
    k_flash<<<g2, 128, 0, stream>>>(Qb, Kb, Vb, Qb);

    // 3) intra output projection -> d_out (64x128 tiles)
    dim3 g3((B_ * L_ / 64) * (D_ / 128), 1, NB_);
    k_gemm_oproj_intra<<<g3, 128, 0, stream>>>(Qb, Wo, bo, out);

    // 4) router QKV (gather last tokens from d_out)
    dim3 g4(D_ / 64, 3, 1);
    k_gemm_qkv_inter<<<g4, 128, 0, stream>>>(out, iWq, ibq, iWk, ibk, iWv, ibv, q2, k2, v2);

    // 5) tiny 4-token attention
    k_inter_attn<<<1, 128, 0, stream>>>(q2, k2, v2, a2);

    // 6) router output projection, scatter into last-token slots of d_out
    dim3 g6(D_ / 64, 1, 1);
    k_gemm_oproj_inter<<<g6, 128, 0, stream>>>(a2, iWo, ibo, out);
}
